// RGAT_90520730730498
// MI455X (gfx1250) — matrix operand order repaired and checked by reading, not compiled, not run on hardware
//
#include <hip/hip_runtime.h>
#include <stdint.h>

#define HIDC 512
#define NEC  5
#define OUTC 153
#define OUTP 192
#define INC  768
#define EPSV 1e-5f

typedef __attribute__((ext_vector_type(16))) __bf16 v16bf;
typedef __attribute__((ext_vector_type(8)))  float  v8f;

union Frag16 { uint4 u[2]; v16bf v; };

__device__ __forceinline__ uint16_t f2bf(float f) {
    unsigned u = __float_as_uint(f);
    u += 0x7FFFu + ((u >> 16) & 1u);          // round-to-nearest-even
    return (uint16_t)(u >> 16);
}
__device__ __forceinline__ float bf2f(uint16_t h) {
    return __uint_as_float(((unsigned)h) << 16);
}

// ---------------------------------------------------------------- converts
__global__ void k_f2bf(const float* __restrict__ s, uint16_t* __restrict__ d, size_t n) {
    size_t i = (size_t)blockIdx.x * blockDim.x + threadIdx.x;
    size_t st = (size_t)gridDim.x * blockDim.x;
    for (; i < n; i += st) d[i] = f2bf(s[i]);
}

// weight convert + transpose: src[K x C] row-major f32 -> dst[C x K] row-major bf16
// (i.e. column-major of original => GEMM B staging becomes contiguous b128 copies)
__global__ void k_f2bf_t(const float* __restrict__ s, uint16_t* __restrict__ d,
                         int K, int C) {
    size_t n = (size_t)K * C;
    size_t i = (size_t)blockIdx.x * blockDim.x + threadIdx.x;
    size_t st = (size_t)gridDim.x * blockDim.x;
    for (; i < n; i += st) {
        int k = (int)(i / C), c = (int)(i % C);
        d[(size_t)c * K + k] = f2bf(s[i]);
    }
}

// mlp_W2: src[512 x 153] f32 -> dst[192 x 512] bf16 transposed, zero-padded cols
__global__ void k_pad_w2_t(const float* __restrict__ s, uint16_t* __restrict__ d) {
    int i = blockIdx.x * blockDim.x + threadIdx.x;
    if (i >= OUTP * HIDC) return;
    int col = i / HIDC, k = i % HIDC;
    d[i] = (col < OUTC) ? f2bf(s[k * OUTC + col]) : (uint16_t)0;
}

__global__ void k_fill(float* __restrict__ p, float v, size_t n) {
    size_t i = (size_t)blockIdx.x * blockDim.x + threadIdx.x;
    size_t st = (size_t)gridDim.x * blockDim.x;
    for (; i < n; i += st) p[i] = v;
}

// ---------------------------------------------------------------- WMMA GEMM
// C[M,Nreal] = A[M,K](bf16 row-major) @ Bt[Npad,K](bf16, pre-transposed) (+bias).
// Block 256 thr = 8 waves; tile 128x128; wave subtile 32x64 (2x4 WMMAs); K step 32.
__global__ __launch_bounds__(256)
void k_gemm_bf16(const uint16_t* __restrict__ A, const uint16_t* __restrict__ Bt,
                 float* __restrict__ Cf, uint16_t* __restrict__ Cb,
                 const float* __restrict__ bias,
                 int M, int K, int Npad, int Nreal, int ldc)
{
    __shared__ __align__(16) uint16_t lsA[128 * 40];   // rows,   stride 40 halves
    __shared__ __align__(16) uint16_t lsB[128 * 40];   // cols,   stride 40 halves

    const int tid  = threadIdx.x;
    const int lane = tid & 31;
    const int wave = tid >> 5;
    const int wr   = wave >> 1;            // 0..3  (32-row subtile)
    const int wc   = wave & 1;             // 0..1  (64-col subtile)
    const int rowBase = blockIdx.y * 128;
    const int colBase = blockIdx.x * 128;

    v8f acc[2][4];
    #pragma unroll
    for (int i = 0; i < 2; i++)
        #pragma unroll
        for (int j = 0; j < 4; j++)
            #pragma unroll
            for (int v = 0; v < 8; v++) acc[i][j][v] = 0.f;

    const uint4* A4 = reinterpret_cast<const uint4*>(lsA);
    const uint4* B4 = reinterpret_cast<const uint4*>(lsB);

    for (int k0 = 0; k0 < K; k0 += 32) {
        // ---- stage A tile 128x32 (2 x uint4 per thread), row-major
        #pragma unroll
        for (int t = 0; t < 2; t++) {
            int vec = tid + t * 256;               // 0..511
            int row = vec >> 2;                    // 0..127
            int kh  = (vec & 3) * 8;               // 0,8,16,24
            int gr  = rowBase + row;
            uint4 d = {0u, 0u, 0u, 0u};
            if (gr < M)
                d = *reinterpret_cast<const uint4*>(A + (size_t)gr * K + k0 + kh);
            *reinterpret_cast<uint4*>(&lsA[row * 40 + kh]) = d;
        }
        // ---- stage B tile 128 cols x 32 k (2 x uint4 per thread), already col-major
        #pragma unroll
        for (int t = 0; t < 2; t++) {
            int vec = tid + t * 256;               // 0..511
            int col = vec >> 2;                    // 0..127
            int kh  = (vec & 3) * 8;               // 0,8,16,24
            int gc  = colBase + col;
            uint4 d = {0u, 0u, 0u, 0u};
            if (gc < Npad)
                d = *reinterpret_cast<const uint4*>(Bt + (size_t)gc * K + k0 + kh);
            *reinterpret_cast<uint4*>(&lsB[col * 40 + kh]) = d;
        }
        __syncthreads();

        // ---- fragments (ISA 7.12.2 layouts) + 8 WMMAs
        Frag16 a[2], b[4];
        #pragma unroll
        for (int i = 0; i < 2; i++) {
            int r  = wr * 32 + i * 16 + (lane & 15);
            int k8 = (lane < 16) ? 0 : 1;          // K half-offset /8
            a[i].u[0] = A4[r * 5 + k8];            // K kh..kh+7
            a[i].u[1] = A4[r * 5 + k8 + 2];        // K kh+16..kh+23
        }
        #pragma unroll
        for (int j = 0; j < 4; j++) {
            int c  = wc * 64 + j * 16 + (lane & 15);
            int k8 = (lane < 16) ? 0 : 1;          // K half-offset /8
            b[j].u[0] = B4[c * 5 + k8];            // K kh..kh+7
            b[j].u[1] = B4[c * 5 + k8 + 2];        // K kh+16..kh+23
        }
        #pragma unroll
        for (int i = 0; i < 2; i++)
            #pragma unroll
            for (int j = 0; j < 4; j++)
                acc[i][j] = __builtin_amdgcn_wmma_f32_16x16x32_bf16(
                    false, a[i].v, false, b[j].v, (short)0, acc[i][j], false, false);
        __syncthreads();
    }

    // ---- store (C/D layout: lane = col, VGPR v = row v / 8+v)
    #pragma unroll
    for (int i = 0; i < 2; i++)
        #pragma unroll
        for (int j = 0; j < 4; j++) {
            int cb = colBase + wc * 64 + j * 16 + (lane & 15);
            int rb = rowBase + wr * 32 + i * 16 + ((lane < 16) ? 0 : 8);
            float bv = (bias && cb < Nreal) ? bias[cb] : 0.f;
            #pragma unroll
            for (int v = 0; v < 8; v++) {
                int r = rb + v;
                if (r < M && cb < Nreal) {
                    float val = acc[i][j][v] + bv;
                    if (Cf) Cf[(size_t)r * ldc + cb] = val;
                    else    Cb[(size_t)r * ldc + cb] = f2bf(val);
                }
            }
        }
}

// ---------------------------------------------------------------- el/er dot
// wave per node: lane l covers cols l*16..l*16+15 (head = l/8), xor-reduce in 8-lane groups
__global__ __launch_bounds__(256)
void k_eler(const uint16_t* __restrict__ f, const float* __restrict__ al,
            const float* __restrict__ ar, float* __restrict__ el,
            float* __restrict__ er, int Nn)
{
    int wave = threadIdx.x >> 5, lane = threadIdx.x & 31;
    int n = blockIdx.x * 8 + wave;
    if (n >= Nn) return;
    const uint16_t* fr = f + (size_t)n * HIDC + lane * 16;
    float pl = 0.f, pr = 0.f;
    #pragma unroll
    for (int q = 0; q < 16; q++) {
        float xv = bf2f(fr[q]);
        pl += xv * al[lane * 16 + q];
        pr += xv * ar[lane * 16 + q];
    }
    pl += __shfl_xor(pl, 1); pr += __shfl_xor(pr, 1);
    pl += __shfl_xor(pl, 2); pr += __shfl_xor(pr, 2);
    pl += __shfl_xor(pl, 4); pr += __shfl_xor(pr, 4);
    if ((lane & 7) == 0) {
        el[n * 4 + (lane >> 3)] = pl;
        er[n * 4 + (lane >> 3)] = pr;
    }
}

// ---------------------------------------------------------------- edge softmax
__device__ __forceinline__ void atomicMaxF(float* a, float v) {
    if (v >= 0.f) atomicMax((int*)a, __float_as_int(v));
    else          atomicMin((unsigned int*)a, __float_as_uint(v));
}

__global__ void k_edge_max(const int* __restrict__ src, const int* __restrict__ dst,
                           const int* __restrict__ et, int j,
                           const float* __restrict__ el, const float* __restrict__ er,
                           float* __restrict__ m, int E4)
{
    int i  = blockIdx.x * blockDim.x + threadIdx.x;
    int st = gridDim.x * blockDim.x;
    for (; i < E4; i += st) {
        int e = i >> 2, h = i & 3;
        if (et[e] != j) continue;
        float sc = el[src[e] * 4 + h] + er[dst[e] * 4 + h];
        sc = (sc > 0.f) ? sc : 0.2f * sc;               // leaky_relu 0.2
        atomicMaxF(&m[dst[e] * 4 + h], sc);
    }
}

__global__ void k_edge_z(const int* __restrict__ src, const int* __restrict__ dst,
                         const int* __restrict__ et, int j,
                         const float* __restrict__ el, const float* __restrict__ er,
                         const float* __restrict__ m, float* __restrict__ z,
                         float* __restrict__ s, int E4)
{
    int i  = blockIdx.x * blockDim.x + threadIdx.x;
    int st = gridDim.x * blockDim.x;
    for (; i < E4; i += st) {
        int e = i >> 2, h = i & 3;
        if (et[e] != j) continue;
        float sc = el[src[e] * 4 + h] + er[dst[e] * 4 + h];
        sc = (sc > 0.f) ? sc : 0.2f * sc;
        float zv = __expf(sc - m[dst[e] * 4 + h]);
        z[i] = zv;
        atomicAdd(&s[dst[e] * 4 + h], zv);
    }
}

// wave per edge; lane l handles 16 cols (head l/8); f gather (L2-resident) + f32 scatter-add
__global__ __launch_bounds__(256)
void k_aggregate(const int* __restrict__ src, const int* __restrict__ dst,
                 const int* __restrict__ et, int j,
                 const uint16_t* __restrict__ f, const float* __restrict__ z,
                 const float* __restrict__ s, float* __restrict__ out, int E)
{
    int wave = threadIdx.x >> 5, lane = threadIdx.x & 31;
    int h = lane >> 3;
    for (int e = blockIdx.x * 8 + wave; e < E; e += gridDim.x * 8) {
        if (et[e] != j) continue;
        int sn = src[e], dn = dst[e];
        float alpha = z[e * 4 + h] / fmaxf(s[dn * 4 + h], 1e-9f);
        const uint16_t* fr = f + (size_t)sn * HIDC + lane * 16;
        float* o = out + (size_t)dn * HIDC + lane * 16;
        uint4 d0 = *reinterpret_cast<const uint4*>(fr);
        uint4 d1 = *reinterpret_cast<const uint4*>(fr + 8);
        const uint16_t* hh = reinterpret_cast<const uint16_t*>(&d0);
        #pragma unroll
        for (int q = 0; q < 8; q++) atomicAdd(&o[q], alpha * bf2f(hh[q]));
        hh = reinterpret_cast<const uint16_t*>(&d1);
        #pragma unroll
        for (int q = 0; q < 8; q++) atomicAdd(&o[8 + q], alpha * bf2f(hh[q]));
    }
}

// ---------------------------------------------------------------- BatchNorm
__global__ __launch_bounds__(512)
void k_colstats(const float* __restrict__ x, float* __restrict__ cs,
                float* __restrict__ cq, int Nn)
{
    int c = threadIdx.x;                               // column 0..511
    float s0 = 0.f, s1 = 0.f;
    for (int r = blockIdx.x; r < Nn; r += gridDim.x) {
        float v = x[(size_t)r * HIDC + c];
        s0 += v; s1 += v * v;
    }
    atomicAdd(&cs[c], s0);
    atomicAdd(&cq[c], s1);
}

__global__ void k_bn_act(float* __restrict__ x, uint16_t* __restrict__ xb,
                         const float* __restrict__ cs, const float* __restrict__ cq,
                         const float* __restrict__ g, const float* __restrict__ b,
                         int Nn, int relu)
{
    size_t n = (size_t)Nn * HIDC;
    size_t i = (size_t)blockIdx.x * blockDim.x + threadIdx.x;
    size_t st = (size_t)gridDim.x * blockDim.x;
    float invN = 1.f / (float)Nn;
    for (; i < n; i += st) {
        int c = (int)(i & (HIDC - 1));
        float mu  = cs[c] * invN;
        float var = cq[c] * invN - mu * mu;
        float y = g[c] * (x[i] - mu) * rsqrtf(var + EPSV) + b[c];
        y = relu ? fmaxf(y, 0.f) : ((y > 0.f) ? y : expm1f(y));
        x[i]  = y;
        xb[i] = f2bf(y);
    }
}

// ---------------------------------------------------------------- launcher
extern "C" void kernel_launch(void* const* d_in, const int* in_sizes, int n_in,
                              void* d_out, int out_size, void* d_ws, size_t ws_size,
                              hipStream_t stream)
{
    (void)n_in; (void)out_size; (void)ws_size;
    const float* x   = (const float*)d_in[0];
    const int*   src = (const int*)d_in[1];
    const int*   dst = (const int*)d_in[2];
    const int*   et  = (const int*)d_in[3];
    const float* W0  = (const float*)d_in[4];
    const float* W1  = (const float*)d_in[5];
    const float* aL  = (const float*)d_in[6];
    const float* aR  = (const float*)d_in[7];
    // d_in[8] conv_bias: zero + per-column constant -> cancelled by BN mean-subtract
    const float* sW0 = (const float*)d_in[9];
    const float* sb0 = (const float*)d_in[10];
    const float* sW1 = (const float*)d_in[11];
    const float* sb1 = (const float*)d_in[12];
    const float* ng  = (const float*)d_in[13];
    const float* nb  = (const float*)d_in[14];
    const float* mW1 = (const float*)d_in[15];
    const float* mb1 = (const float*)d_in[16];
    const float* mg  = (const float*)d_in[17];
    const float* mbt = (const float*)d_in[18];
    const float* mW2 = (const float*)d_in[19];
    const float* mb2 = (const float*)d_in[20];

    const int Nn = in_sizes[0] / INC;
    const int E  = in_sizes[1];

    // bump allocator over workspace (total ~480 MB; per-etype f kept at 102 MB for L2 residency)
    char* base = (char*)d_ws;
    size_t off = 0;
    auto alloc = [&](size_t bytes) -> void* {
        void* p = base + off;
        off = (off + bytes + 255) & ~(size_t)255;
        return p;
    };
    uint16_t* hb = (uint16_t*)alloc((size_t)Nn * INC * 2);   // bf16 activations
    float*    xs = (float*)   alloc((size_t)Nn * HIDC * 4);  // skip accumulator / h / t1
    uint16_t* fB = (uint16_t*)alloc((size_t)Nn * HIDC * 2);  // per-etype features (bf16)
    uint16_t* Wb = (uint16_t*)alloc((size_t)INC * HIDC * 2); // bf16 weights, transposed (reused)
    float*    el = (float*)alloc((size_t)Nn * 4 * 4);
    float*    er = (float*)alloc((size_t)Nn * 4 * 4);
    float*    mB = (float*)alloc((size_t)Nn * 4 * 4);
    float*    sB = (float*)alloc((size_t)Nn * 4 * 4);
    float*    zB = (float*)alloc((size_t)E * 4 * 4);
    float*    cs = (float*)alloc(HIDC * 4);
    float*    cq = (float*)alloc(HIDC * 4);

    const int CT = 256;
    dim3 gg(HIDC / 128, (Nn + 127) / 128);

    for (int L = 0; L < 2; L++) {
        int K = L ? HIDC : INC;
        const float* hsrc = L ? xs : x;
        const float* skW  = L ? sW1 : sW0;
        const float* skb  = L ? sb1 : sb0;
        const float* Wl   = L ? W1 : W0;

        k_f2bf<<<4096, CT, 0, stream>>>(hsrc, hb, (size_t)Nn * K);
        k_f2bf_t<<<2048, CT, 0, stream>>>(skW, Wb, K, HIDC);
        k_gemm_bf16<<<gg, 256, 0, stream>>>(hb, Wb, xs, nullptr, skb,
                                            Nn, K, HIDC, HIDC, HIDC);
        for (int j = 0; j < NEC; j++) {
            k_f2bf_t<<<2048, CT, 0, stream>>>(Wl + (size_t)j * K * HIDC, Wb, K, HIDC);
            k_gemm_bf16<<<gg, 256, 0, stream>>>(hb, Wb, nullptr, fB, nullptr,
                                                Nn, K, HIDC, HIDC, HIDC);
            k_eler<<<(Nn + 7) / 8, 256, 0, stream>>>(
                fB, aL + (size_t)(L * NEC + j) * HIDC,
                aR + (size_t)(L * NEC + j) * HIDC, el, er, Nn);
            k_fill<<<1024, CT, 0, stream>>>(mB, -1e30f, (size_t)Nn * 4);
            k_fill<<<1024, CT, 0, stream>>>(sB, 0.f, (size_t)Nn * 4);
            k_edge_max<<<4096, CT, 0, stream>>>(src, dst, et, j, el, er, mB, E * 4);
            k_edge_z  <<<4096, CT, 0, stream>>>(src, dst, et, j, el, er, mB, zB, sB, E * 4);
            k_aggregate<<<4096, 256, 0, stream>>>(src, dst, et, j, fB, zB, sB, xs, E);
        }
        k_fill<<<4, CT, 0, stream>>>(cs, 0.f, HIDC);
        k_fill<<<4, CT, 0, stream>>>(cq, 0.f, HIDC);
        k_colstats<<<1024, 512, 0, stream>>>(xs, cs, cq, Nn);
        k_bn_act<<<4096, CT, 0, stream>>>(xs, hb, cs, cq,
                                          ng + L * HIDC, nb + L * HIDC, Nn, 0);
    }

    // MLP head: t1 = relu(bn(h @ W1 + b1)); out = t1 @ W2 + b2
    k_f2bf_t<<<2048, CT, 0, stream>>>(mW1, Wb, HIDC, HIDC);
    k_gemm_bf16<<<gg, 256, 0, stream>>>(hb, Wb, xs, nullptr, mb1,
                                        Nn, HIDC, HIDC, HIDC, HIDC);
    k_fill<<<4, CT, 0, stream>>>(cs, 0.f, HIDC);
    k_fill<<<4, CT, 0, stream>>>(cq, 0.f, HIDC);
    k_colstats<<<1024, 512, 0, stream>>>(xs, cs, cq, Nn);
    k_bn_act<<<4096, CT, 0, stream>>>(xs, hb, cs, cq, mg, mbt, Nn, 1);
    k_pad_w2_t<<<(OUTP * HIDC + CT - 1) / CT, CT, 0, stream>>>(mW2, Wb);
    dim3 go((OUTP + 127) / 128, (Nn + 127) / 128);
    k_gemm_bf16<<<go, 256, 0, stream>>>(hb, Wb, (float*)d_out, nullptr, mb2,
                                        Nn, HIDC, OUTP, OUTC, OUTC);
}
